// Qwen_53317724013009
// MI455X (gfx1250) — compile-verified, matrix operand order checked by reading, not run on hardware
//
#include <hip/hip_runtime.h>
#include <hip/hip_bf16.h>

typedef __bf16 bf16;
typedef __attribute__((ext_vector_type(16))) __bf16 v16bf;
typedef __attribute__((ext_vector_type(8)))  float  v8f;
typedef __attribute__((ext_vector_type(4)))  unsigned int u32x4;
typedef __attribute__((ext_vector_type(8)))  int i32x8;
typedef __attribute__((ext_vector_type(4)))  int i32x4;

#define B_  2
#define S_  1024
#define D_  2048
#define H_  16
#define KV_ 4
#define HD_ 128
#define RD_ 64
#define E_  8
#define MI_ 768
#define SI_ 4096
#define T_  (B_ * S_)
#define EPS_ 1e-6f

// ---------------------------------------------------------------------------
// WMMA fragment helpers (16-bit A/B operand layout, CDNA5 ISA 7.12.2):
//   lanes 0-15 : row m0+lane, K = k0+0..7  and k0+16..23
//   lanes 16-31: row m0+lane-16, K = k0+8..15 and k0+24..31
// Two 16B loads per fragment -> ds_load_b128 / global_load_b128.
// ---------------------------------------------------------------------------
union FragU { struct { uint4 lo, hi; } u; v16bf v; };

__device__ __forceinline__ v16bf frag_load(const bf16* mat, int ld, int m0, int k0) {
  const int lane = threadIdx.x & 31;
  const int half = lane >> 4;
  const bf16* p = mat + (size_t)(m0 + (lane & 15)) * ld + k0 + half * 8;
  FragU f;
  f.u.lo = *(const uint4*)(p);
  f.u.hi = *(const uint4*)(p + 16);
  return f.v;
}

#define LDT 48  // padded LDS row stride (bf16 elems): 96B, multiple of 16B

// ---------------------------------------------------------------------------
// Tensor Data Mover: DMA one (128 rows x 32 cols) bf16 tile, row stride K
// elems in global, into LDS with row stride LDT (=48) via the TDM pad
// feature: pad_interval = 16 dwords (one 64B tile row), pad_amount = 8 dwords
// (32B pad -> 48 bf16 LDS stride). Descriptor per CDNA5 ISA 8.3/8.4.
// Issue from one wave only (TDM ignores EXEC; one issue per wave).
// This toolchain exposes the 6-arg builtin: (g0, g1, g2, g3, g4, cpol).
// ---------------------------------------------------------------------------
__device__ __forceinline__ void tdm_load_tile(const bf16* gptr, unsigned lds_off, int K) {
  unsigned long long ga = (unsigned long long)(uintptr_t)gptr;
  u32x4 g0 = {
      1u,                                              // count=1, user D#
      lds_off,                                         // lds_addr (bytes)
      (unsigned)(ga & 0xffffffffu),                    // global_addr[31:0]
      (unsigned)((ga >> 32) & 0x01ffffffu) | 0x80000000u  // addr[56:32] | type=2
  };
  i32x8 g1 = {
      (int)(0x00010000u            // data_size = 1 (2 bytes)
            | 0x00100000u          // pad_enable
            | (3u << 22)           // pad_interval: 16 dwords
            | (7u << 25)),         // pad_amount: 8 dwords
      (int)((unsigned)(K & 0xffff) << 16),   // tensor_dim0[15:0] = K
      (int)((unsigned)K >> 16),              // tensor_dim0[31:16], dim1[15:0]=0
      (int)(0x10u | (32u << 16)),            // tensor_dim1[31:16] (=1M rows), tile_dim0=32
      (int)128,                              // tile_dim1 = 128, tile_dim2 = 0
      (int)K,                                // tensor_dim0_stride[31:0] = K elems
      0, 0                                   // stride hi, dim1_stride (unused, 2D)
  };
  i32x4 z4 = {0, 0, 0, 0};
  i32x8 z8 = {0, 0, 0, 0, 0, 0, 0, 0};
  __builtin_amdgcn_tensor_load_to_lds(g0, g1, z4, z4, z8, 0);
}

// ---------------------------------------------------------------------------
// Generic GEMM: C[M,N] (f32) = A[M,K] (bf16, row-major) * Bt[N,K]^T (bf16)
// 128x128 block tile, BK=32, 256 threads = 8 waves, wave tile 64x32 (4x2 WMMA)
// TDM double-buffered LDS staging tracked on TENSORcnt.
// ---------------------------------------------------------------------------
__global__ __launch_bounds__(256) void qwen_gemm_bf16(
    const bf16* __restrict__ A, const bf16* __restrict__ Bt, float* __restrict__ C,
    int M, int N, int K) {
  __shared__ __align__(16) bf16 sA[2][128 * LDT];
  __shared__ __align__(16) bf16 sB[2][128 * LDT];
  const int tid  = threadIdx.x;
  const int lane = tid & 31;
  const int wave = tid >> 5;
  const int wr = wave >> 2, wc = wave & 3;
  const int bm = blockIdx.y * 128, bn = blockIdx.x * 128;
  const int ktiles = K >> 5;
  v8f acc[4][2] = {};

  if (wave == 0) {  // issue first tile pair
    tdm_load_tile(&A [(size_t)bm * K], (unsigned)(uintptr_t)&sA[0][0], K);
    tdm_load_tile(&Bt[(size_t)bn * K], (unsigned)(uintptr_t)&sB[0][0], K);
  }
  for (int t = 0; t < ktiles; ++t) {
    const int buf = t & 1;
    if (wave == 0) {
      if (t + 1 < ktiles) {  // issue next pair into other buffer, wait for current
        tdm_load_tile(&A [(size_t)bm * K + (t + 1) * 32], (unsigned)(uintptr_t)&sA[buf ^ 1][0], K);
        tdm_load_tile(&Bt[(size_t)bn * K + (t + 1) * 32], (unsigned)(uintptr_t)&sB[buf ^ 1][0], K);
        __builtin_amdgcn_s_wait_tensorcnt(2);
      } else {
        __builtin_amdgcn_s_wait_tensorcnt(0);
      }
    }
    __syncthreads();  // data-arrived handoff from wave 0 to all waves
    v16bf aF[4], bF[2];
    #pragma unroll
    for (int i = 0; i < 4; ++i) aF[i] = frag_load(&sA[buf][0], LDT, wr * 64 + i * 16, 0);
    #pragma unroll
    for (int j = 0; j < 2; ++j) bF[j] = frag_load(&sB[buf][0], LDT, wc * 32 + j * 16, 0);
    #pragma unroll
    for (int i = 0; i < 4; ++i)
      #pragma unroll
      for (int j = 0; j < 2; ++j)
        acc[i][j] = __builtin_amdgcn_wmma_f32_16x16x32_bf16(
            false, aF[i], false, bF[j], (short)0, acc[i][j], false, false);
    __syncthreads();  // all reads of buf done before it is refilled
  }
  // epilogue: C layout VGPR r -> (M=r | r+8), N = lane&15
  const int half = lane >> 4, ln = lane & 15;
  #pragma unroll
  for (int i = 0; i < 4; ++i)
    #pragma unroll
    for (int j = 0; j < 2; ++j) {
      int row0 = bm + wr * 64 + i * 16 + half * 8;
      int col  = bn + wc * 32 + j * 16 + ln;
      #pragma unroll
      for (int r = 0; r < 8; ++r)
        C[(size_t)(row0 + r) * N + col] = acc[i][j][r];
    }
}

// ---------------------------------------------------------------------------
// Flash attention: block = (qblock of 64, head, batch), 4 waves x 16 q rows.
// q,k,v bf16; online softmax; causal mask analytic. o: f32 (B,S,H,HD)
// ---------------------------------------------------------------------------
__global__ __launch_bounds__(128) void qwen_attn(
    const bf16* __restrict__ q, const bf16* __restrict__ kx,
    const bf16* __restrict__ vx, float* __restrict__ o) {
  const int b = blockIdx.z, h = blockIdx.y, qb = blockIdx.x;
  const int kvh = h / (H_ / KV_);
  const int wave = threadIdx.x >> 5, lane = threadIdx.x & 31;
  const int half = lane >> 4, ln = lane & 15;
  const int q0 = qb * 64 + wave * 16;

  __shared__ __align__(16) bf16 sK [32 * 136];   // 32 kv rows x HD (pad 8)
  __shared__ __align__(16) bf16 sVt[128 * LDT];  // HD rows x 32 kv (transposed)
  __shared__ __align__(16) bf16 sP [4][16 * LDT];

  // Q fragments straight from global (A-operand layout)
  v16bf aQ[4];
  {
    const bf16* qrow = q + ((((size_t)b * S_) + q0 + ln) * H_ + h) * HD_;
    #pragma unroll
    for (int c = 0; c < 4; ++c) {
      FragU f;
      f.u.lo = *(const uint4*)(qrow + c * 32 + half * 8);
      f.u.hi = *(const uint4*)(qrow + c * 32 + half * 8 + 16);
      aQ[c] = f.v;
    }
  }
  v8f accO[8] = {};
  float rm[8], rl[8];
  #pragma unroll
  for (int r = 0; r < 8; ++r) { rm[r] = -1e30f; rl[r] = 0.f; }
  const float scale = 0.08838834764831845f;  // 1/sqrt(HD)
  const int nchunks = qb * 2 + 2;            // causal: kv up to block end

  for (int kc = 0; kc < nchunks; ++kc) {
    const int kbase = kc * 32;
    // cooperative load: K chunk row-major, V chunk transposed
    #pragma unroll
    for (int i = 0; i < 2; ++i) {
      int idx = threadIdx.x + i * 128;       // 0..255 (uint4), 8 per row
      int r = idx >> 3;
      int c8 = (idx & 7) * 8;
      size_t gro = ((((size_t)b * S_) + kbase + r) * KV_ + kvh) * HD_ + c8;
      *(uint4*)&sK[r * 136 + c8] = *(const uint4*)&kx[gro];
      uint4 vv = *(const uint4*)&vx[gro];
      const bf16* ve = (const bf16*)&vv;
      #pragma unroll
      for (int e = 0; e < 8; ++e) sVt[(c8 + e) * LDT + r] = ve[e];
      if (kc + 1 < nchunks) {  // CDNA5 prefetch of next kv chunk
        __builtin_prefetch(&kx[gro + (size_t)32 * KV_ * HD_], 0, 0);
        __builtin_prefetch(&vx[gro + (size_t)32 * KV_ * HD_], 0, 0);
      }
    }
    __syncthreads();
    if (kbase <= q0 + 15) {  // wave-uniform
      v8f sS[2];
      #pragma unroll
      for (int j = 0; j < 2; ++j) {
        v8f s = {};
        #pragma unroll
        for (int c = 0; c < 4; ++c) {
          v16bf bK = frag_load(sK, 136, j * 16, c * 32);
          s = __builtin_amdgcn_wmma_f32_16x16x32_bf16(
              false, aQ[c], false, bK, (short)0, s, false, false);
        }
        #pragma unroll
        for (int r = 0; r < 8; ++r) {
          float sv = s[r] * scale;
          int kpos = kbase + j * 16 + ln;
          int qpos = q0 + r + half * 8;
          s[r] = (kpos > qpos) ? -1e30f : sv;
        }
        sS[j] = s;
      }
      #pragma unroll
      for (int r = 0; r < 8; ++r) {
        float m = fmaxf(sS[0][r], sS[1][r]);
        #pragma unroll
        for (int off = 8; off >= 1; off >>= 1) m = fmaxf(m, __shfl_xor(m, off));
        float newm = fmaxf(rm[r], m);
        float f  = __expf(rm[r] - newm);
        float p0 = __expf(sS[0][r] - newm);
        float p1 = __expf(sS[1][r] - newm);
        float rs = p0 + p1;
        #pragma unroll
        for (int off = 8; off >= 1; off >>= 1) rs += __shfl_xor(rs, off);
        rl[r] = rl[r] * f + rs;
        rm[r] = newm;
        #pragma unroll
        for (int t = 0; t < 8; ++t) accO[t][r] *= f;
        sS[0][r] = p0; sS[1][r] = p1;
      }
      // P (16x32) -> LDS bf16, reload as A fragment
      bf16* pb = &sP[wave][0];
      #pragma unroll
      for (int j = 0; j < 2; ++j)
        #pragma unroll
        for (int r = 0; r < 8; ++r)
          pb[(r + half * 8) * LDT + j * 16 + ln] = (bf16)sS[j][r];
      v16bf aP = frag_load(pb, LDT, 0, 0);
      #pragma unroll
      for (int t = 0; t < 8; ++t) {
        v16bf bV = frag_load(sVt, LDT, t * 16, 0);
        accO[t] = __builtin_amdgcn_wmma_f32_16x16x32_bf16(
            false, aP, false, bV, (short)0, accO[t], false, false);
      }
    }
    __syncthreads();
  }
  float* orow = o + ((((size_t)b * S_) + q0) * H_ + h) * HD_;
  #pragma unroll
  for (int r = 0; r < 8; ++r) {
    float inv = 1.0f / rl[r];
    int row = r + half * 8;
    #pragma unroll
    for (int t = 0; t < 8; ++t)
      orow[(size_t)row * H_ * HD_ + t * 16 + ln] = accO[t][r] * inv;
  }
}

// ---------------------------------------------------------------------------
// Small fused kernels
// ---------------------------------------------------------------------------
__global__ __launch_bounds__(256) void qwen_rmsnorm_bf16(
    const float* __restrict__ in, const float* __restrict__ w,
    bf16* __restrict__ out, int ncols) {
  __shared__ float red[8];
  const int row = blockIdx.x;
  const float* x = in + (size_t)row * ncols;
  float ss = 0.f;
  for (int c = threadIdx.x; c < ncols; c += 256) { float v = x[c]; ss += v * v; }
  for (int off = 16; off >= 1; off >>= 1) ss += __shfl_xor(ss, off);
  if ((threadIdx.x & 31) == 0) red[threadIdx.x >> 5] = ss;
  __syncthreads();
  if (threadIdx.x == 0) {
    float t = 0; for (int i = 0; i < 8; ++i) t += red[i];
    red[0] = rsqrtf(t / ncols + EPS_);
  }
  __syncthreads();
  float rn = red[0];
  bf16* y = out + (size_t)row * ncols;
  for (int c = threadIdx.x; c < ncols; c += 256)
    y[c] = (bf16)(x[c] * rn * (1.0f + w[c]));
}

// RMS (over HD) + RoPE (first RD dims), f32 in -> bf16 out. 128 thr = 1 head row
__global__ __launch_bounds__(128) void qwen_rmsrope_bf16(
    const float* __restrict__ in, int head_stride, const float* __restrict__ nw,
    const float* __restrict__ fc, bf16* __restrict__ out, int nh) {
  __shared__ float buf[HD_];
  __shared__ float red[4];
  const int head = blockIdx.x % nh;
  const size_t token = blockIdx.x / nh;
  const int s = (int)(token % S_);
  const float* x = in + (token * nh + head) * (size_t)head_stride;
  int d = threadIdx.x;
  float v = x[d];
  float ss = v * v;
  for (int off = 16; off >= 1; off >>= 1) ss += __shfl_xor(ss, off);
  if ((d & 31) == 0) red[d >> 5] = ss;
  __syncthreads();
  if (d == 0) red[0] = rsqrtf((red[0] + red[1] + red[2] + red[3]) / HD_ + EPS_);
  __syncthreads();
  v = v * red[0] * (1.0f + nw[d]);
  buf[d] = v;
  __syncthreads();
  float ov = v;
  if (d < RD_) {
    int i = d >> 1;
    float c  = fc[((size_t)s * (RD_ / 2) + i) * 2 + 0];
    float sn = fc[((size_t)s * (RD_ / 2) + i) * 2 + 1];
    float x1 = buf[i * 2], x2 = buf[i * 2 + 1];
    ov = (d & 1) ? (x1 * sn + x2 * c) : (x1 * c - x2 * sn);
  }
  out[(token * nh + head) * (size_t)HD_ + d] = (bf16)ov;
}

__global__ __launch_bounds__(256) void qwen_cvt_bf16(
    const float* __restrict__ in, bf16* __restrict__ out, size_t n) {
  for (size_t i = blockIdx.x * (size_t)blockDim.x + threadIdx.x; i < n;
       i += (size_t)gridDim.x * blockDim.x)
    out[i] = (bf16)in[i];
}

// f32 (rows,cols) -> bf16 (cols,rows)
__global__ __launch_bounds__(256) void qwen_transpose_cvt(
    const float* __restrict__ in, bf16* __restrict__ out, int rows, int cols) {
  __shared__ float tile[32][33];
  const int tx = threadIdx.x & 31, ty = threadIdx.x >> 5;
  const int c0 = blockIdx.x * 32, r0 = blockIdx.y * 32;
  #pragma unroll
  for (int i = 0; i < 32; i += 8) {
    int r = r0 + ty + i, c = c0 + tx;
    tile[ty + i][tx] = (r < rows && c < cols) ? in[(size_t)r * cols + c] : 0.f;
  }
  __syncthreads();
  #pragma unroll
  for (int i = 0; i < 32; i += 8) {
    int r = c0 + ty + i, c = r0 + tx;
    if (r < cols && c < rows) out[(size_t)r * rows + c] = (bf16)tile[tx][ty + i];
  }
}

__global__ __launch_bounds__(256) void qwen_gate_o(
    const float* __restrict__ o, const float* __restrict__ xq,
    bf16* __restrict__ og, size_t n) {
  for (size_t i = blockIdx.x * (size_t)blockDim.x + threadIdx.x; i < n;
       i += (size_t)gridDim.x * blockDim.x) {
    size_t t = i / (H_ * HD_);
    int rem = (int)(i % (H_ * HD_));
    int h = rem / HD_, d = rem % HD_;
    float g = xq[(t * H_ + h) * (size_t)(2 * HD_) + HD_ + d];
    og[i] = (bf16)(o[i] * (1.f / (1.f + __expf(-g))));
  }
}

__global__ __launch_bounds__(256) void qwen_add(
    const float* __restrict__ a, const float* __restrict__ b,
    float* __restrict__ c, size_t n) {
  for (size_t i = blockIdx.x * (size_t)blockDim.x + threadIdx.x; i < n;
       i += (size_t)gridDim.x * blockDim.x)
    c[i] = a[i] + b[i];
}

__global__ __launch_bounds__(256) void qwen_router(
    const bf16* __restrict__ t, const float* __restrict__ rw,
    float* __restrict__ wfull) {
  __shared__ float wred[8];
  __shared__ float logits[E_];
  const int tok = blockIdx.x;
  for (int e = 0; e < E_; ++e) {
    float acc = 0.f;
    for (int d = threadIdx.x; d < D_; d += 256)
      acc += (float)t[(size_t)tok * D_ + d] * rw[(size_t)e * D_ + d];
    for (int off = 16; off >= 1; off >>= 1) acc += __shfl_xor(acc, off);
    if ((threadIdx.x & 31) == 0) wred[threadIdx.x >> 5] = acc;
    __syncthreads();
    if (threadIdx.x == 0) {
      float s = 0; for (int i = 0; i < 8; ++i) s += wred[i];
      logits[e] = s;
    }
    __syncthreads();
  }
  if (threadIdx.x == 0) {
    float mx = logits[0];
    for (int e = 1; e < E_; ++e) mx = fmaxf(mx, logits[e]);
    float p[E_]; float se = 0.f;
    for (int e = 0; e < E_; ++e) { p[e] = __expf(logits[e] - mx); se += p[e]; }
    for (int e = 0; e < E_; ++e) p[e] /= se;
    int i0 = 0;
    for (int e = 1; e < E_; ++e) if (p[e] > p[i0]) i0 = e;
    int i1 = (i0 == 0) ? 1 : 0;
    for (int e = 0; e < E_; ++e) if (e != i0 && p[e] > p[i1]) i1 = e;
    float s2 = p[i0] + p[i1];
    for (int e = 0; e < E_; ++e) wfull[(size_t)tok * E_ + e] = 0.f;
    wfull[(size_t)tok * E_ + i0] = p[i0] / s2;
    wfull[(size_t)tok * E_ + i1] = p[i1] / s2;
  }
}

__global__ __launch_bounds__(256) void qwen_silu_mul(
    const float* __restrict__ g, const float* __restrict__ u,
    bf16* __restrict__ out, int rowlen, int gstride, size_t n) {
  for (size_t i = blockIdx.x * (size_t)blockDim.x + threadIdx.x; i < n;
       i += (size_t)gridDim.x * blockDim.x) {
    size_t t = i / rowlen;
    int j = (int)(i % rowlen);
    float gv = g[t * (size_t)gstride + j];
    float uv = u[t * (size_t)gstride + j];
    out[i] = (bf16)(gv / (1.f + __expf(-gv)) * uv);
  }
}

__global__ __launch_bounds__(256) void qwen_moe_accum(
    float* __restrict__ moe, const float* __restrict__ ye,
    const float* __restrict__ wfull, int e, int first, size_t n) {
  for (size_t i = blockIdx.x * (size_t)blockDim.x + threadIdx.x; i < n;
       i += (size_t)gridDim.x * blockDim.x) {
    size_t t = i / D_;
    float v = wfull[t * E_ + e] * ye[i];
    moe[i] = first ? v : (moe[i] + v);
  }
}

__global__ __launch_bounds__(256) void qwen_sgate(
    const bf16* __restrict__ t, const float* __restrict__ w,
    float* __restrict__ sg) {
  __shared__ float red[8];
  const int tok = blockIdx.x;
  float acc = 0.f;
  for (int d = threadIdx.x; d < D_; d += 256)
    acc += (float)t[(size_t)tok * D_ + d] * w[d];
  for (int off = 16; off >= 1; off >>= 1) acc += __shfl_xor(acc, off);
  if ((threadIdx.x & 31) == 0) red[threadIdx.x >> 5] = acc;
  __syncthreads();
  if (threadIdx.x == 0) {
    float s = 0; for (int i = 0; i < 8; ++i) s += red[i];
    sg[tok] = 1.f / (1.f + __expf(-s));
  }
}

__global__ __launch_bounds__(256) void qwen_final(
    const float* __restrict__ xmid, const float* __restrict__ moe,
    const float* __restrict__ sh, const float* __restrict__ sg,
    float* __restrict__ out, size_t n) {
  for (size_t i = blockIdx.x * (size_t)blockDim.x + threadIdx.x; i < n;
       i += (size_t)gridDim.x * blockDim.x) {
    size_t t = i / D_;
    out[i] = xmid[i] + moe[i] + sg[t] * sh[i];
  }
}

// ---------------------------------------------------------------------------
// Workspace layout (lifetime-overlapped arenas, peak ~196 MB)
// ---------------------------------------------------------------------------
constexpr size_t OFF_XQ    = 0;                         // xq f32 -> g f32 -> sh f32 (33.5MB)
constexpr size_t OFF_O     = OFF_XQ + 33554432;         // o f32 (lo half) -> u f32 (33.5MB)
constexpr size_t OFF_OPROJ = OFF_O + 16777216;          // oproj f32 (hi half of o/u arena)
constexpr size_t OFF_WQT   = OFF_O + 33554432;          // wqT -> sgT -> act_sh (16.8MB)
constexpr size_t OFF_WOT   = OFF_WQT + 16777216;        // woT(8.4) -> suT -> sdT (16.8MB)
constexpr size_t OFF_OG    = OFF_WOT + 8388608;         // og bf16 (8.4MB, with woT)
constexpr size_t OFF_H     = OFF_WOT + 16777216;        // h bf16 -> t bf16 (8.4MB)
constexpr size_t OFF_KBT   = OFF_H + 8388608;           // wkT -> k bf16 (2.1MB)
constexpr size_t OFF_VBT   = OFF_KBT + 2097152;         // wvT -> v bf16 (2.1MB)
constexpr size_t OFF_KF    = OFF_VBT + 2097152;         // k f32 (4.2MB)
constexpr size_t OFF_VF    = OFF_KF + 4194304;          // v f32 (4.2MB)
constexpr size_t OFF_QB    = OFF_VF + 4194304;          // q bf16 (8.4MB)
constexpr size_t OFF_XMID  = OFF_QB + 8388608;          // 16.8MB
constexpr size_t OFF_MOE   = OFF_XMID + 16777216;       // 16.8MB
constexpr size_t OFF_WFULL = OFF_MOE + 16777216;        // 64KB
constexpr size_t OFF_SGATE = OFF_WFULL + 65536;         // 64KB
constexpr size_t OFF_WET   = OFF_SGATE + 65536;         // expert gu weight bf16 (6.3MB)
constexpr size_t OFF_GU    = OFF_WET + 6291456;         // gu f32 (12.6MB)
constexpr size_t OFF_ACT   = OFF_GU + 12582912;         // act bf16 (3.1MB)
constexpr size_t OFF_WDT   = OFF_ACT + 3145728;         // expert down weight bf16 (3.1MB)
constexpr size_t OFF_YE    = OFF_WDT + 3145728;         // ye f32 (16.8MB)

extern "C" void kernel_launch(void* const* d_in, const int* in_sizes, int n_in,
                              void* d_out, int out_size, void* d_ws, size_t ws_size,
                              hipStream_t stream) {
  (void)in_sizes; (void)n_in; (void)out_size; (void)ws_size;
  const float* x   = (const float*)d_in[0];
  const float* fc  = (const float*)d_in[1];
  /* d_in[2] mask: causal, computed analytically */
  const float* wq  = (const float*)d_in[3];
  const float* wk  = (const float*)d_in[4];
  const float* wv  = (const float*)d_in[5];
  const float* wo  = (const float*)d_in[6];
  const float* qn  = (const float*)d_in[7];
  const float* kn  = (const float*)d_in[8];
  const float* anw = (const float*)d_in[9];
  const float* fnw = (const float*)d_in[10];
  const float* rw  = (const float*)d_in[11];
  const float* egu = (const float*)d_in[12];
  const float* edn = (const float*)d_in[13];
  const float* sgw = (const float*)d_in[14];
  const float* suw = (const float*)d_in[15];
  const float* sdw = (const float*)d_in[16];
  const float* seg = (const float*)d_in[17];
  float* out = (float*)d_out;
  char* ws = (char*)d_ws;

  float* xq_f   = (float*)(ws + OFF_XQ);
  float* o_f    = (float*)(ws + OFF_O);
  float* oproj  = (float*)(ws + OFF_OPROJ);
  float* u_f    = (float*)(ws + OFF_O);      // alias after attention done
  float* g_f    = (float*)(ws + OFF_XQ);     // alias after gate consumed
  float* sh_f   = (float*)(ws + OFF_XQ);     // alias after act_sh computed
  bf16* wqT     = (bf16*)(ws + OFF_WQT);
  bf16* sgT     = (bf16*)(ws + OFF_WQT);
  bf16* actsh   = (bf16*)(ws + OFF_WQT);
  bf16* woT     = (bf16*)(ws + OFF_WOT);
  bf16* suT     = (bf16*)(ws + OFF_WOT);
  bf16* sdT     = (bf16*)(ws + OFF_WOT);
  bf16* og_b    = (bf16*)(ws + OFF_OG);
  bf16* h_b     = (bf16*)(ws + OFF_H);
  bf16* t_b     = (bf16*)(ws + OFF_H);
  bf16* wkT     = (bf16*)(ws + OFF_KBT);
  bf16* k_b     = (bf16*)(ws + OFF_KBT);
  bf16* wvT     = (bf16*)(ws + OFF_VBT);
  bf16* v_b     = (bf16*)(ws + OFF_VBT);
  float* k_f    = (float*)(ws + OFF_KF);
  float* v_f    = (float*)(ws + OFF_VF);
  bf16* q_b     = (bf16*)(ws + OFF_QB);
  float* xmid   = (float*)(ws + OFF_XMID);
  float* moe    = (float*)(ws + OFF_MOE);
  float* wfull  = (float*)(ws + OFF_WFULL);
  float* sgate  = (float*)(ws + OFF_SGATE);
  bf16* weT     = (bf16*)(ws + OFF_WET);
  float* gu_f   = (float*)(ws + OFF_GU);
  bf16* act_b   = (bf16*)(ws + OFF_ACT);
  bf16* wdT     = (bf16*)(ws + OFF_WDT);
  float* ye_f   = (float*)(ws + OFF_YE);

  const size_t nTD = (size_t)T_ * D_;

  // ---- attention branch ----
  qwen_rmsnorm_bf16<<<T_, 256, 0, stream>>>(x, anw, h_b, D_);
  qwen_transpose_cvt<<<dim3(4096 / 32, 2048 / 32), 256, 0, stream>>>(wq, wqT, D_, H_ * HD_ * 2);
  qwen_transpose_cvt<<<dim3(512 / 32, 2048 / 32), 256, 0, stream>>>(wk, wkT, D_, KV_ * HD_);
  qwen_transpose_cvt<<<dim3(512 / 32, 2048 / 32), 256, 0, stream>>>(wv, wvT, D_, KV_ * HD_);
  qwen_gemm_bf16<<<dim3(4096 / 128, T_ / 128), 256, 0, stream>>>(h_b, wqT, xq_f, T_, H_ * HD_ * 2, D_);
  qwen_gemm_bf16<<<dim3(512 / 128, T_ / 128), 256, 0, stream>>>(h_b, wkT, k_f, T_, KV_ * HD_, D_);
  qwen_gemm_bf16<<<dim3(512 / 128, T_ / 128), 256, 0, stream>>>(h_b, wvT, v_f, T_, KV_ * HD_, D_);
  qwen_rmsrope_bf16<<<T_ * H_, 128, 0, stream>>>(xq_f, 2 * HD_, qn, fc, q_b, H_);
  qwen_rmsrope_bf16<<<T_ * KV_, 128, 0, stream>>>(k_f, HD_, kn, fc, k_b, KV_);
  qwen_cvt_bf16<<<2048, 256, 0, stream>>>(v_f, v_b, (size_t)T_ * KV_ * HD_);
  qwen_attn<<<dim3(S_ / 64, H_, B_), 128, 0, stream>>>(q_b, k_b, v_b, o_f);
  qwen_gate_o<<<4096, 256, 0, stream>>>(o_f, xq_f, og_b, (size_t)T_ * H_ * HD_);
  qwen_transpose_cvt<<<dim3(2048 / 32, 2048 / 32), 256, 0, stream>>>(wo, woT, H_ * HD_, D_);
  qwen_gemm_bf16<<<dim3(D_ / 128, T_ / 128), 256, 0, stream>>>(og_b, woT, oproj, T_, D_, H_ * HD_);
  qwen_add<<<4096, 256, 0, stream>>>(x, oproj, xmid, nTD);

  // ---- MoE branch ----
  qwen_rmsnorm_bf16<<<T_, 256, 0, stream>>>(xmid, fnw, t_b, D_);
  qwen_router<<<T_, 256, 0, stream>>>(t_b, rw, wfull);
  for (int e = 0; e < E_; ++e) {
    qwen_cvt_bf16<<<4096, 256, 0, stream>>>(egu + (size_t)e * 2 * MI_ * D_, weT, (size_t)2 * MI_ * D_);
    qwen_gemm_bf16<<<dim3(2 * MI_ / 128, T_ / 128), 256, 0, stream>>>(t_b, weT, gu_f, T_, 2 * MI_, D_);
    qwen_silu_mul<<<4096, 256, 0, stream>>>(gu_f, gu_f + MI_, act_b, MI_, 2 * MI_, (size_t)T_ * MI_);
    qwen_cvt_bf16<<<4096, 256, 0, stream>>>(edn + (size_t)e * D_ * MI_, wdT, (size_t)D_ * MI_);
    qwen_gemm_bf16<<<dim3(D_ / 128, T_ / 128), 256, 0, stream>>>(act_b, wdT, ye_f, T_, D_, MI_);
    qwen_moe_accum<<<4096, 256, 0, stream>>>(moe, ye_f, wfull, e, (e == 0) ? 1 : 0, nTD);
  }

  // ---- shared expert ----
  qwen_transpose_cvt<<<dim3(SI_ / 32, D_ / 32), 256, 0, stream>>>(sgw, sgT, D_, SI_);
  qwen_gemm_bf16<<<dim3(SI_ / 128, T_ / 128), 256, 0, stream>>>(t_b, sgT, g_f, T_, SI_, D_);
  qwen_transpose_cvt<<<dim3(SI_ / 32, D_ / 32), 256, 0, stream>>>(suw, suT, D_, SI_);
  qwen_gemm_bf16<<<dim3(SI_ / 128, T_ / 128), 256, 0, stream>>>(t_b, suT, u_f, T_, SI_, D_);
  qwen_silu_mul<<<8192, 256, 0, stream>>>(g_f, u_f, actsh, SI_, SI_, (size_t)T_ * SI_);
  qwen_transpose_cvt<<<dim3(D_ / 32, SI_ / 32), 256, 0, stream>>>(sdw, sdT, SI_, D_);
  qwen_gemm_bf16<<<dim3(D_ / 128, T_ / 128), 256, 0, stream>>>(actsh, sdT, sh_f, T_, D_, SI_);
  qwen_sgate<<<T_, 256, 0, stream>>>(t_b, seg, sgate);

  // ---- combine ----
  qwen_final<<<4096, 256, 0, stream>>>(xmid, moe, sh_f, sgate, out, nTD);
}
